// MultiHeadSelfAttention_38242388803623
// MI455X (gfx1250) — matrix operand order repaired and checked by reading, not compiled, not run on hardware
//
#include <hip/hip_runtime.h>

// ---------------------------------------------------------------------------
// MI455X (gfx1250) multi-head self-attention, bf16 WMMA pipeline.
//   ~137 GFLOP vs ~80MB HBM traffic -> compute bound -> all GEMMs on
//   v_wmma_f32_16x16x32_bf16.  Intermediates (~92MB bf16) live in workspace
//   and fit in the 192MB L2.  Weight tiles are staged into LDS with
//   double-buffered async copies (global_load_async_to_lds_b128 + ASYNCcnt)
//   and shared by the 4 waves of each block.
// ---------------------------------------------------------------------------

typedef unsigned int  u32;
typedef unsigned short u16;
typedef __attribute__((ext_vector_type(16))) __bf16 bf16x16;
typedef __attribute__((ext_vector_type(8)))  float  v8f;
typedef __attribute__((ext_vector_type(4)))  int    v4i;

union AF { bf16x16 v; u32 u[8]; };

#define S_LEN   2048
#define DMODEL  1024
#define NHEADS  16
#define HD      64

#if defined(__has_builtin)
#  if __has_builtin(__builtin_amdgcn_global_load_async_to_lds_b128)
#    define HAVE_ASYNC_LDS 1
#  endif
#endif
#ifndef HAVE_ASYNC_LDS
#  define HAVE_ASYNC_LDS 0
#endif

__device__ __forceinline__ u16 f2bf(float f) {          // fp32 -> bf16 RNE
  u32 u = __float_as_uint(f);
  u = u + 0x7FFFu + ((u >> 16) & 1u);
  return (u16)(u >> 16);
}

__device__ __forceinline__ v8f v8f_zero() {
  v8f z = {0.f,0.f,0.f,0.f,0.f,0.f,0.f,0.f};
  return z;
}

__device__ __forceinline__ v8f wmma_bf16(const AF& a, const AF& b, v8f c) {
  return __builtin_amdgcn_wmma_f32_16x16x32_bf16(false, a.v, false, b.v,
                                                 (short)0, c, false, false);
}

__device__ __forceinline__ void wait_lds() {
#if __has_builtin(__builtin_amdgcn_s_wait_dscnt)
  __builtin_amdgcn_s_wait_dscnt(0);
#else
  asm volatile("s_wait_dscnt 0" ::: "memory");
#endif
}

// 16-byte async copy global -> LDS (per lane).  Falls back to a synchronous
// b128 load + ds_store_b128 when the async builtin is unavailable.
__device__ __forceinline__ void cp_async16(u16* lds_dst, const u16* gsrc) {
#if HAVE_ASYNC_LDS
  __builtin_amdgcn_global_load_async_to_lds_b128(
      (__attribute__((address_space(1))) v4i*)gsrc,
      (__attribute__((address_space(3))) v4i*)lds_dst,
      0, 0);
#else
  *(uint4*)lds_dst = *(const uint4*)gsrc;
#endif
}

__device__ __forceinline__ void wait_async_all() {
#if HAVE_ASYNC_LDS
#  if __has_builtin(__builtin_amdgcn_s_wait_asynccnt)
  __builtin_amdgcn_s_wait_asynccnt(0);
#  else
  asm volatile("s_wait_asynccnt 0x0" ::: "memory");
#  endif
#else
  wait_lds();
#endif
}

// A-fragment K offset for bf16 16x32 A matrix (ISA 7.12.2)
__device__ __forceinline__ int a_koff(int v, int g) {
  return ((v >> 2) << 4) + (g << 3) + ((v & 3) << 1);
}
// B-fragment K offset for bf16 32x16 B matrix
__device__ __forceinline__ int b_koff(int v, int g) {
  return ((v >> 2) << 4) + (g << 3) + ((v & 3) << 1);
}

// Stage one 64(N) x 32(K) bf16 weight tile into LDS: 128 threads x 32B.
__device__ __forceinline__ void stage_B(u16 (*Bs)[32], const u16* __restrict__ wb,
                                        int e0, int kk, int tid) {
  const int row  = tid >> 1;
  const int half = tid & 1;
  const u16* src = wb + (size_t)(e0 + row) * DMODEL + kk + (half << 4);
  u16* dst = &Bs[row][half << 4];
  cp_async16(dst, src);
  cp_async16(dst + 8, src + 8);
}

// ---------------------------------------------------------------------------
__global__ void cvt_bf16_kernel(const float* __restrict__ src,
                                u16* __restrict__ dst, int n) {
  int i = blockIdx.x * blockDim.x + threadIdx.x;
  int stride = gridDim.x * blockDim.x;
  for (; i < n; i += stride) dst[i] = f2bf(src[i]);
}

// ---------------------------------------------------------------------------
// qkv = x @ Wqkv^T  (M=8192, N=3072, K=1024), fused RoPE + scatter.
// Block: 4 waves = 64(M) x 64(N); B tile shared via async-LDS double buffer.
__global__ void __launch_bounds__(128) qkv_rope_kernel(
    const u16* __restrict__ xb, const u16* __restrict__ wb,
    const int* __restrict__ tok,
    u16* __restrict__ qo, u16* __restrict__ ko, u16* __restrict__ vto)
{
  __shared__ __align__(16) u16 Bs[2][64][32];
  const int tid  = threadIdx.x;
  const int lane = tid & 31;
  const int wv   = tid >> 5;
  const int mb = blockIdx.x & 127;        // 128 M-blocks of 64 rows
  const int nt = blockIdx.x >> 7;         // 48 N tiles
  const int m0 = (mb * 4 + wv) << 4;
  const int e0 = nt << 6;
  const int g = lane >> 4, ln = lane & 15;

  v8f acc[4];
  #pragma unroll
  for (int i = 0; i < 4; i++) acc[i] = v8f_zero();

  const u16* arow = xb + (size_t)(m0 + ln) * DMODEL;

  stage_B(Bs[0], wb, e0, 0, tid);
  for (int kk = 0, it = 0; kk < DMODEL; kk += 32, ++it) {
    const int buf = it & 1;
    wait_async_all();          // own async copies for Bs[buf] done
    __syncthreads();           // all waves' copies done; prev reads done
    if (kk + 32 < DMODEL) stage_B(Bs[buf ^ 1], wb, e0, kk + 32, tid);

    AF a;
    #pragma unroll
    for (int v = 0; v < 8; v++)
      a.u[v] = *(const u32*)(arow + kk + a_koff(v, g));
    #pragma unroll
    for (int ns = 0; ns < 4; ns++) {
      AF b;
      #pragma unroll
      for (int v = 0; v < 8; v++)
        b.u[v] = *(const u32*)(&Bs[buf][(ns << 4) + ln][b_koff(v, g)]);
      acc[ns] = wmma_bf16(a, b, acc[ns]);
    }
  }

  #pragma unroll
  for (int ns = 0; ns < 4; ns++) {
    const int e = e0 + (ns << 4) + ln;      // feature column, fixed per lane
    const int which = e >> 10;              // 0=q 1=k 2=v
    const int r = e & 1023;
    const int h = r >> 6;
    const int d = r & 63;
    if (which == 2) {
      #pragma unroll
      for (int v = 0; v < 8; v++) {
        int t = m0 + v + (g << 3);
        int bb = t >> 11, s = t & (S_LEN - 1);
        vto[(((size_t)bb * NHEADS + h) * HD + d) * S_LEN + s] = f2bf(acc[ns][v]);
      }
    } else {
      // RoPE: pair (2i, 2i+1); partner value lives in adjacent lane.
      const float invf = __expf(-0.28782313662425576f * (float)(d >> 1));
      u16* dst = (which == 0) ? qo : ko;
      const float qs = (which == 0) ? 0.125f : 1.0f;   // 1/sqrt(64) pre-scale
      #pragma unroll
      for (int v = 0; v < 8; v++) {
        int t = m0 + v + (g << 3);
        int bb = t >> 11, s = t & (S_LEN - 1);
        float val = acc[ns][v];
        float partner = __shfl_xor(val, 1, 32);
        float ang = (float)tok[s] * invf;
        float sn, cs; __sincosf(ang, &sn, &cs);
        float o = val * cs + ((d & 1) ? partner * sn : -partner * sn);
        dst[(((size_t)bb * NHEADS + h) * S_LEN + s) * HD + d] = f2bf(o * qs);
      }
    }
  }
}

// ---------------------------------------------------------------------------
// Flash attention: each wave owns 16 query rows of one (b,h).
// Per 32-key step: 4 WMMAs for S=Q*K^T, online softmax (16-lane shuffles),
// P staged through LDS (C-layout -> A-layout re-stripe), 4 WMMAs for P*V.
__global__ void __launch_bounds__(128) flash_attn_kernel(
    const u16* __restrict__ qb, const u16* __restrict__ kb,
    const u16* __restrict__ vtb, u16* __restrict__ attnb)
{
  __shared__ __align__(16) u16 Pl[4][16][32];   // per-wave 16x32 bf16 P tile
  const int lane = threadIdx.x & 31;
  const int wv = threadIdx.x >> 5;
  const int wid = blockIdx.x * 4 + wv;
  const int bh = wid >> 7;           // 64 (b,h) pairs
  const int q0 = (wid & 127) << 4;   // 128 query tiles of 16
  const int g = lane >> 4, ln = lane & 15;

  const u16* qp = qb  + (size_t)bh * S_LEN * HD;
  const u16* kp = kb  + (size_t)bh * S_LEN * HD;
  const u16* vp = vtb + (size_t)bh * HD * S_LEN;

  AF aQ0, aQ1;                       // Q 16x64, split in two K=32 fragments
  {
    const u16* r = qp + (size_t)(q0 + ln) * HD;
    #pragma unroll
    for (int v = 0; v < 8; v++) {
      int K = a_koff(v, g);
      aQ0.u[v] = *(const u32*)(r + K);
      aQ1.u[v] = *(const u32*)(r + 32 + K);
    }
  }

  float m[8], l[8];
  v8f O[4];
  #pragma unroll
  for (int v = 0; v < 8; v++) { m[v] = -__builtin_inff(); l[v] = 0.f; }
  #pragma unroll
  for (int t = 0; t < 4; t++) O[t] = v8f_zero();

  for (int j0 = 0; j0 <= q0 + 15; j0 += 32) {
    float sc[2][8];
    #pragma unroll
    for (int nh = 0; nh < 2; nh++) {
      int jr = j0 + (nh << 4) + ln;
      int jc = jr < S_LEN ? jr : (S_LEN - 1);   // clamp: masked later anyway
      const u16* kr = kp + (size_t)jc * HD;
      AF b0, b1;
      #pragma unroll
      for (int v = 0; v < 8; v++) {
        int K = b_koff(v, g);
        b0.u[v] = *(const u32*)(kr + K);
        b1.u[v] = *(const u32*)(kr + 32 + K);
      }
      v8f sa = v8f_zero();
      sa = wmma_bf16(aQ0, b0, sa);
      sa = wmma_bf16(aQ1, b1, sa);
      #pragma unroll
      for (int v = 0; v < 8; v++) sc[nh][v] = sa[v];
    }

    #pragma unroll
    for (int v = 0; v < 8; v++) {
      int qrow = q0 + v + (g << 3);
      float s0 = (j0 + ln       <= qrow) ? sc[0][v] : -__builtin_inff();
      float s1 = (j0 + 16 + ln  <= qrow) ? sc[1][v] : -__builtin_inff();
      float rm = fmaxf(s0, s1);
      #pragma unroll
      for (int off = 1; off < 16; off <<= 1)
        rm = fmaxf(rm, __shfl_xor(rm, off, 32));
      float mn = fmaxf(m[v], rm);
      float al = __expf(m[v] - mn);
      float p0 = __expf(s0 - mn);
      float p1 = __expf(s1 - mn);
      float rs = p0 + p1;
      #pragma unroll
      for (int off = 1; off < 16; off <<= 1)
        rs += __shfl_xor(rs, off, 32);
      l[v] = l[v] * al + rs;
      m[v] = mn;
      #pragma unroll
      for (int t = 0; t < 4; t++) O[t][v] *= al;
      int row = v + (g << 3);
      Pl[wv][row][ln]      = f2bf(p0);
      Pl[wv][row][16 + ln] = f2bf(p1);
    }
    wait_lds();                      // wave-private LDS RAW

    AF aP;
    {
      const u16* pr = &Pl[wv][ln][0];
      #pragma unroll
      for (int v = 0; v < 8; v++)
        aP.u[v] = *(const u32*)(pr + a_koff(v, g));
    }
    #pragma unroll
    for (int t = 0; t < 4; t++) {
      const u16* vr = vp + (size_t)((t << 4) + ln) * S_LEN;
      AF bV;
      #pragma unroll
      for (int v = 0; v < 8; v++) {
        int jj = j0 + b_koff(v, g);
        if (jj > S_LEN - 2) jj = S_LEN - 2;  // keep operands finite (p==0 there)
        bV.u[v] = *(const u32*)(vr + jj);
      }
      O[t] = wmma_bf16(aP, bV, O[t]);
    }
  }

  const int bb = bh >> 4, h = bh & 15;
  #pragma unroll
  for (int t = 0; t < 4; t++) {
    #pragma unroll
    for (int v = 0; v < 8; v++) {
      int srow = q0 + v + (g << 3);
      int col = h * HD + (t << 4) + ln;
      float val = O[t][v] / l[v];
      attnb[(size_t)(bb * S_LEN + srow) * DMODEL + col] = f2bf(val);
    }
  }
}

// ---------------------------------------------------------------------------
// out = attn @ Wout^T  (M=8192, N=1024, K=1024), f32 stores to d_out.
// Same async-LDS B-tile staging as the QKV GEMM.
__global__ void __launch_bounds__(128) out_proj_kernel(
    const u16* __restrict__ ab, const u16* __restrict__ wb,
    float* __restrict__ out)
{
  __shared__ __align__(16) u16 Bs[2][64][32];
  const int tid  = threadIdx.x;
  const int lane = tid & 31;
  const int wv   = tid >> 5;
  const int mb = blockIdx.x & 127;        // 128 M-blocks of 64 rows
  const int nt = blockIdx.x >> 7;         // 16 N tiles
  const int m0 = (mb * 4 + wv) << 4;
  const int e0 = nt << 6;
  const int g = lane >> 4, ln = lane & 15;

  v8f acc[4];
  #pragma unroll
  for (int i = 0; i < 4; i++) acc[i] = v8f_zero();

  const u16* arow = ab + (size_t)(m0 + ln) * DMODEL;

  stage_B(Bs[0], wb, e0, 0, tid);
  for (int kk = 0, it = 0; kk < DMODEL; kk += 32, ++it) {
    const int buf = it & 1;
    wait_async_all();
    __syncthreads();
    if (kk + 32 < DMODEL) stage_B(Bs[buf ^ 1], wb, e0, kk + 32, tid);

    AF a;
    #pragma unroll
    for (int v = 0; v < 8; v++)
      a.u[v] = *(const u32*)(arow + kk + a_koff(v, g));
    #pragma unroll
    for (int ns = 0; ns < 4; ns++) {
      AF b;
      #pragma unroll
      for (int v = 0; v < 8; v++)
        b.u[v] = *(const u32*)(&Bs[buf][(ns << 4) + ln][b_koff(v, g)]);
      acc[ns] = wmma_bf16(a, b, acc[ns]);
    }
  }
  #pragma unroll
  for (int ns = 0; ns < 4; ns++) {
    int e = e0 + (ns << 4) + ln;
    #pragma unroll
    for (int v = 0; v < 8; v++) {
      int t = m0 + v + (g << 3);
      out[(size_t)t * DMODEL + e] = acc[ns][v];
    }
  }
}

// ---------------------------------------------------------------------------
extern "C" void kernel_launch(void* const* d_in, const int* in_sizes, int n_in,
                              void* d_out, int out_size, void* d_ws, size_t ws_size,
                              hipStream_t stream)
{
  (void)in_sizes; (void)n_in; (void)out_size; (void)ws_size;
  const float* x    = (const float*)d_in[0];
  const int*   tok  = (const int*)  d_in[1];
  const float* Wqkv = (const float*)d_in[2];
  const float* Wout = (const float*)d_in[3];
  float* out = (float*)d_out;

  const size_t NX  = (size_t)8192 * 1024;        // x elements
  const size_t NWQ = (size_t)3072 * 1024;        // W_qkv
  const size_t NWO = (size_t)1024 * 1024;        // W_out
  const size_t NQK = (size_t)4 * 16 * 2048 * 64; // per q/k/vt/attn tensor

  u16* xb    = (u16*)d_ws;
  u16* wqkvb = xb    + NX;
  u16* woutb = wqkvb + NWQ;
  u16* qb    = woutb + NWO;
  u16* kb    = qb    + NQK;
  u16* vtb   = kb    + NQK;
  u16* attnb = vtb   + NQK;

  cvt_bf16_kernel<<<2048, 256, 0, stream>>>(x,    xb,    (int)NX);
  cvt_bf16_kernel<<<1024, 256, 0, stream>>>(Wqkv, wqkvb, (int)NWQ);
  cvt_bf16_kernel<<<512,  256, 0, stream>>>(Wout, woutb, (int)NWO);

  // 128 M-blocks x 48 N-tiles, 4 waves/block (64x64 per block)
  qkv_rope_kernel<<<6144, 128, 0, stream>>>(xb, wqkvb, tok, qb, kb, vtb);
  // 64 (b,h) x 128 q-tiles = 8192 waves
  flash_attn_kernel<<<2048, 128, 0, stream>>>(qb, kb, vtb, attnb);
  // 128 M-blocks x 16 N-tiles
  out_proj_kernel<<<2048, 128, 0, stream>>>(attnb, woutb, out);
}
